// CrossAttention_66571993088906
// MI455X (gfx1250) — matrix operand order repaired and checked by reading, not compiled, not run on hardware
//
#include <hip/hip_runtime.h>
#include <hip/hip_bf16.h>
#include <math.h>

// Problem constants (match reference)
#define BB   4
#define CC   64
#define NHD  4
#define DK   16
#define NN   4096          // 64*64 spatial tokens
// dk^-0.5 * log2(e): softmax computed in exp2 domain
#define QSCALE_F (0.25f * 1.44269504088896340736f)
#define EPS_F   1e-5f
#define BN_CNT  16384.0f   // B*H*W = 4*4096

typedef __attribute__((ext_vector_type(16))) _Float16 v16h;
typedef __attribute__((ext_vector_type(8)))  _Float16 v8h;
typedef __attribute__((ext_vector_type(2)))  __fp16   v2fp;   // cvt_pkrtz result type
typedef __attribute__((ext_vector_type(8)))  float    v8f;
typedef __attribute__((ext_vector_type(4)))  float    v4f;
typedef __attribute__((ext_vector_type(2)))  float    v2f;

// ---------------------------------------------------------------------------
// Phase 0: zero the BN stats accumulator (2 dirs * 64 ch * {sum,sumsq})
// ---------------------------------------------------------------------------
__global__ void zero_stats_kernel(float* stats) {
    if (threadIdx.x < 2 * CC * 2) stats[threadIdx.x] = 0.0f;
}

// ---------------------------------------------------------------------------
// Phase 1: QKV 1x1 conv for both images.
//   Q (scaled by log2(e)/sqrt(dk)) -> f16 [img][b][h][n][16]  (WMMA A side)
//   K                              -> f16 [img][b][h][n][16]  (WMMA B side)
//   V transposed                   -> f16 [img][b][h][d][N]   (WMMA B side, PV)
// ---------------------------------------------------------------------------
__global__ void qkv_kernel(const float* __restrict__ img1,
                           const float* __restrict__ img2,
                           const float* __restrict__ qkv_w,
                           _Float16* __restrict__ Qs,
                           _Float16* __restrict__ Kf,
                           _Float16* __restrict__ Vt) {
    // grid.x = 2*B*192*(N/256) = 24576, block 256
    const int n    = (int)threadIdx.x + (int)(blockIdx.x & 15) * 256;
    const int o    = ((int)blockIdx.x >> 4) % 192;
    const int ib   = ((int)blockIdx.x >> 4) / 192;   // 0..7
    const int b    = ib & 3;
    const int im   = ib >> 2;

    const float* x = (im ? img2 : img1) + (size_t)(b * CC) * NN + n;
    const float* w = qkv_w + (size_t)o * CC;

    float acc = 0.0f;
#pragma unroll
    for (int c = 0; c < CC; ++c) acc = fmaf(w[c], x[(size_t)c * NN], acc);

    const int h = o / 48;
    const int g = o % 48;
    const size_t headbase = ((size_t)(im * BB + b) * NHD + h);
    if (g < 16) {
        Qs[(headbase * NN + n) * DK + g] = (_Float16)(acc * QSCALE_F);
    } else if (g < 32) {
        Kf[(headbase * NN + n) * DK + (g - 16)] = (_Float16)acc;
    } else {
        Vt[(headbase * DK + (g - 32)) * (size_t)NN + n] = (_Float16)acc;
    }
}

// ---------------------------------------------------------------------------
// Phase 2: cross flash-attention with v_wmma_f32_16x16x32_f16.
// One wave owns a 16-query tile; 8 waves / 256-thread block.
// Per 32-key step: 2 score WMMAs (K padded 16->32), LDS transpose of raw f32
// scores into row-per-lane domain, in-register softmax (1 shfl_xor for max,
// 1 for sum, raw v_exp_f32), f16 P packed via cvt_pkrtz straight into the
// A operand, 1 full-K PV WMMA.
// ---------------------------------------------------------------------------
__global__ void __launch_bounds__(256)
flash_kernel(const _Float16* __restrict__ Qs,
             const _Float16* __restrict__ Kf,
             const _Float16* __restrict__ Vt,
             float* __restrict__ Oatt) {
    // Interleaved score tile: Slds[w][row][2*col + s] = score(row, col + 16*s)
    __shared__ __align__(16) float Slds[8][16][32];
    __shared__ __align__(16) float Alds[8][16];      // per-row alpha / 1/l bcast

    const int lane = (int)threadIdx.x & 31;
    const int wave = (int)threadIdx.x >> 5;
    int tile = (int)blockIdx.x * 8 + wave;               // 0..8191
    const int qt  = tile & 255;  tile >>= 8;             // 256 query tiles
    const int h   = tile & 3;    tile >>= 2;
    const int b   = tile & 3;    tile >>= 2;
    const int dir = tile & 1;                            // 0: Q1·K2V2, 1: Q2·K1V1
    const int src = dir ^ 1;

    const _Float16* qbase = Qs + (((size_t)(dir * BB + b) * NHD + h) * NN + (size_t)qt * 16) * DK;
    const _Float16* kbase = Kf + ((size_t)(src * BB + b) * NHD + h) * NN * DK;
    const _Float16* vbase = Vt + ((size_t)(src * BB + b) * NHD + h) * DK * (size_t)NN;

    const int m  = lane & 15;      // row (A,softmax) / column (B,C) index
    const int hi = lane >> 4;      // wave half

    // Q in 16-bit A layout, K-dim padded 16 -> 32 with zeros.
    v16h aQ = {};
    if (hi == 0) {
        aQ = *(const v16h*)(qbase + (size_t)m * DK);
    }

    v8f acc = {};                  // O accumulator (C layout, f32)
    float mrow = -INFINITY;        // running row max (row m, exp2 domain)
    float lrow = 0.0f;             // running row sum

    for (int k0 = 0; k0 < NN; k0 += 32) {
        // K^T in B layout: lane = key column, lanes 0-15 carry d 0..15, upper K half zero.
        v16h bK0 = {}, bK1 = {};
        if (hi == 0) {
            bK0 = *(const v16h*)(kbase + (size_t)(k0 + m)      * DK);
            bK1 = *(const v16h*)(kbase + (size_t)(k0 + 16 + m) * DK);
        }
        v8f S0 = {}, S1 = {};
        S0 = __builtin_amdgcn_wmma_f32_16x16x32_f16(false, aQ, false, bK0, (short)0, S0, false, false);
        S1 = __builtin_amdgcn_wmma_f32_16x16x32_f16(false, aQ, false, bK1, (short)0, S1, false, false);

        // Scatter C-layout scores to LDS, interleaved pairs -> b64 stores.
#pragma unroll
        for (int r = 0; r < 8; ++r) {
            v2f s2 = { S0[r], S1[r] };
            *(v2f*)&Slds[wave][r + hi * 8][2 * m] = s2;
        }
        asm volatile("s_wait_dscnt 0" ::: "memory");

        // Row-per-lane reload: 16 contiguous floats = keys {8hi..8hi+7, 16+8hi..}.
        float buf[16];
        {
            v4f f0 = *(const v4f*)&Slds[wave][m][16 * hi + 0];
            v4f f1 = *(const v4f*)&Slds[wave][m][16 * hi + 4];
            v4f f2 = *(const v4f*)&Slds[wave][m][16 * hi + 8];
            v4f f3 = *(const v4f*)&Slds[wave][m][16 * hi + 12];
#pragma unroll
            for (int i = 0; i < 4; ++i) {
                buf[i] = f0[i]; buf[4 + i] = f1[i]; buf[8 + i] = f2[i]; buf[12 + i] = f3[i];
            }
        }

        // Row max: pairwise tree (-> v_max3 friendly) + one cross-half shuffle.
        float mx[8];
#pragma unroll
        for (int t = 0; t < 8; ++t) mx[t] = fmaxf(buf[2 * t], buf[2 * t + 1]);
#pragma unroll
        for (int t = 0; t < 4; ++t) mx[t] = fmaxf(mx[t], mx[t + 4]);
        float vmax = fmaxf(fmaxf(mx[0], mx[1]), fmaxf(mx[2], mx[3]));
        vmax = fmaxf(vmax, __shfl_xor(vmax, 16, 32));
        const float mnew = fmaxf(mrow, vmax);

        // Raw v_exp_f32 exponentials + pairwise sum tree.
        float p[16];
#pragma unroll
        for (int t = 0; t < 16; ++t) p[t] = __builtin_amdgcn_exp2f(buf[t] - mnew);
        float sm[8];
#pragma unroll
        for (int t = 0; t < 8; ++t) sm[t] = p[2 * t] + p[2 * t + 1];
#pragma unroll
        for (int t = 0; t < 4; ++t) sm[t] = sm[t] + sm[t + 4];
        float lsum = (sm[0] + sm[1]) + (sm[2] + sm[3]);
        lsum += __shfl_xor(lsum, 16, 32);

        const float alpha = __builtin_amdgcn_exp2f(mrow - mnew);
        lrow = lrow * alpha + lsum;
        mrow = mnew;

        // P in A layout via packed converts: aP[2j,2j+1] = p[4j], p[4j+2]
        // (s=0 keys in elements 0-7), aP[8+2j,9+2j] = p[4j+1], p[4j+3] (s=1).
        union { v16h v; v2fp h2[8]; } up;
#pragma unroll
        for (int j = 0; j < 4; ++j) {
            up.h2[j]     = __builtin_amdgcn_cvt_pkrtz(p[4 * j],     p[4 * j + 2]);
            up.h2[4 + j] = __builtin_amdgcn_cvt_pkrtz(p[4 * j + 1], p[4 * j + 3]);
        }
        v16h aP = up.v;

        // Broadcast alpha (row-per-lane) to the C-layout accumulator rows.
        Alds[wave][m] = alpha;     // lanes m and m+16 write identical value
        asm volatile("s_wait_dscnt 0" ::: "memory");
        {
            v4f a0 = *(const v4f*)&Alds[wave][8 * hi + 0];
            v4f a1 = *(const v4f*)&Alds[wave][8 * hi + 4];
#pragma unroll
            for (int r = 0; r < 4; ++r) { acc[r] *= a0[r]; acc[4 + r] *= a1[r]; }
        }
        asm volatile("" ::: "memory");   // keep next iter's LDS stores after loads

        // V in B layout (full K=32): lane = d column, keys {8hi..8hi+7, 16+8hi..16+8hi+7}.
        v16h bV;
        {
            v8h v0 = *(const v8h*)(vbase + (size_t)m * NN + k0 + hi * 8);
            v8h v1 = *(const v8h*)(vbase + (size_t)m * NN + k0 + 16 + hi * 8);
#pragma unroll
            for (int i = 0; i < 8; ++i) { bV[i] = v0[i]; bV[8 + i] = v1[i]; }
        }
        acc = __builtin_amdgcn_wmma_f32_16x16x32_f16(false, aP, false, bV, (short)0, acc, false, false);
    }

    // Final 1/l redistribution (same LDS broadcast path, once).
    Alds[wave][m] = __builtin_amdgcn_rcpf(lrow);
    asm volatile("s_wait_dscnt 0" ::: "memory");
    float linv[8];
    {
        v4f a0 = *(const v4f*)&Alds[wave][8 * hi + 0];
        v4f a1 = *(const v4f*)&Alds[wave][8 * hi + 4];
#pragma unroll
        for (int r = 0; r < 4; ++r) { linv[r] = a0[r]; linv[4 + r] = a1[r]; }
    }

    // Write O in [dir][b][c=h*16+d][n] layout (8 consecutive n per lane).
    float* obase = Oatt + (((size_t)(dir * BB + b) * CC + h * DK + m) * NN) + (size_t)qt * 16 + hi * 8;
#pragma unroll
    for (int r = 0; r < 8; ++r) obase[r] = acc[r] * linv[r];
}

// ---------------------------------------------------------------------------
// Phase 3a: output projection + bias + residual, plus per-channel sum/sumsq.
// ---------------------------------------------------------------------------
__global__ void proj_stats_kernel(const float* __restrict__ Oatt,
                                  const float* __restrict__ img1,
                                  const float* __restrict__ img2,
                                  const float* __restrict__ out_w,
                                  const float* __restrict__ out_b,
                                  float* __restrict__ Y,
                                  float* __restrict__ stats) {
    // grid.x = 2*B*64*(N/256) = 8192, block 256
    const int tid = (int)threadIdx.x;
    const int n   = tid + (int)(blockIdx.x & 15) * 256;
    const int co  = ((int)blockIdx.x >> 4) & 63;
    const int rest = (int)blockIdx.x >> 10;
    const int b   = rest & 3;
    const int dir = rest >> 2;

    const float* obase = Oatt + (size_t)(dir * BB + b) * CC * NN + n;
    const float* w     = out_w + (size_t)co * CC;
    float acc = out_b[co];
#pragma unroll
    for (int c = 0; c < CC; ++c) acc = fmaf(w[c], obase[(size_t)c * NN], acc);

    const float* img = dir ? img2 : img1;
    acc += img[((size_t)b * CC + co) * NN + n];
    Y[((size_t)(dir * BB + b) * CC + co) * NN + n] = acc;

    __shared__ float s1[256], s2[256];
    s1[tid] = acc;
    s2[tid] = acc * acc;
    __syncthreads();
#pragma unroll
    for (int s = 128; s > 0; s >>= 1) {
        if (tid < s) { s1[tid] += s1[tid + s]; s2[tid] += s2[tid + s]; }
        __syncthreads();
    }
    if (tid == 0) {
        atomicAdd(&stats[(dir * CC + co) * 2 + 0], s1[0]);
        atomicAdd(&stats[(dir * CC + co) * 2 + 1], s2[0]);
    }
}

// ---------------------------------------------------------------------------
// Phase 3b: BatchNorm finalize (batch statistics, biased variance).
// ---------------------------------------------------------------------------
__global__ void bn_kernel(const float* __restrict__ Y,
                          const float* __restrict__ stats,
                          const float* __restrict__ gamma,
                          const float* __restrict__ beta,
                          float* __restrict__ out) {
    const size_t idx = (size_t)blockIdx.x * 256 + threadIdx.x;  // [dir][b][co][n]
    const int co  = (int)((idx / NN) % CC);
    const int dir = (int)(idx / ((size_t)NN * CC * BB));
    const float s1 = stats[(dir * CC + co) * 2 + 0];
    const float s2 = stats[(dir * CC + co) * 2 + 1];
    const float mu  = s1 * (1.0f / BN_CNT);
    const float var = s2 * (1.0f / BN_CNT) - mu * mu;
    const float x = Y[idx];
    out[idx] = (x - mu) * rsqrtf(var + EPS_F) * gamma[co] + beta[co];
}

// ---------------------------------------------------------------------------
extern "C" void kernel_launch(void* const* d_in, const int* in_sizes, int n_in,
                              void* d_out, int out_size, void* d_ws, size_t ws_size,
                              hipStream_t stream) {
    const float* img1  = (const float*)d_in[0];
    const float* img2  = (const float*)d_in[1];
    const float* qkv_w = (const float*)d_in[2];
    const float* out_w = (const float*)d_in[3];
    const float* out_b = (const float*)d_in[4];
    const float* gamma = (const float*)d_in[5];
    const float* beta  = (const float*)d_in[6];
    float* out = (float*)d_out;

    // Workspace carve-up (~22 MB):
    char* ws = (char*)d_ws;
    _Float16* Qs   = (_Float16*)(ws);                       // 2 MB
    _Float16* Kf   = (_Float16*)(ws + ( 2ull << 20));       // 2 MB
    _Float16* Vt   = (_Float16*)(ws + ( 4ull << 20));       // 2 MB
    float*    Oatt = (float*)   (ws + ( 6ull << 20));       // 8 MB
    float*    Y    = (float*)   (ws + (14ull << 20));       // 8 MB
    float*    stats= (float*)   (ws + (22ull << 20));       // 1 KB

    zero_stats_kernel<<<1, 256, 0, stream>>>(stats);
    qkv_kernel<<<2 * BB * 192 * (NN / 256), 256, 0, stream>>>(img1, img2, qkv_w, Qs, Kf, Vt);
    flash_kernel<<<(2 * BB * NHD * (NN / 16)) / 8, 256, 0, stream>>>(Qs, Kf, Vt, Oatt);
    proj_stats_kernel<<<2 * BB * CC * (NN / 256), 256, 0, stream>>>(Oatt, img1, img2, out_w, out_b, Y, stats);
    bn_kernel<<<(2u * BB * CC * NN) / 256, 256, 0, stream>>>(Y, stats, gamma, beta, out);
}
